// HyperbolicLayer_37924561224182
// MI455X (gfx1250) — matrix operand order repaired and checked by reading, not compiled, not run on hardware
//
#include <hip/hip_runtime.h>
#include <hip/hip_bf16.h>
#include <math.h>

typedef __attribute__((ext_vector_type(16))) __bf16 v16bf;
typedef __attribute__((ext_vector_type(8)))  __bf16 v8bf;
typedef __attribute__((ext_vector_type(4)))  __bf16 v4bf;
typedef __attribute__((ext_vector_type(8)))  float  v8f;
typedef __attribute__((ext_vector_type(4)))  float  v4f;
typedef __attribute__((ext_vector_type(4)))  unsigned int v4u;
typedef __attribute__((ext_vector_type(8)))  int    v8i;
typedef __attribute__((ext_vector_type(4)))  int    v4i;

#define EPSV   1e-6f
#define SQRT_C 0.1f     // sqrt(0.01)

// ---------------------------------------------------------------------------
// Pre-pass: convert W (256x256 f32) to bf16 in workspace (row-major, packed).
// ---------------------------------------------------------------------------
__global__ __launch_bounds__(256)
void convert_w_kernel(const float* __restrict__ W, __bf16* __restrict__ Wb)
{
    const int i = (blockIdx.x * 256 + threadIdx.x) * 4;   // 65536 elems total
    v4f w = *(const v4f*)(W + i);
    v4bf t;
    t[0] = (__bf16)w.x; t[1] = (__bf16)w.y;
    t[2] = (__bf16)w.z; t[3] = (__bf16)w.w;
    *(v4bf*)(Wb + i) = t;
}

// ---------------------------------------------------------------------------
// Fused hyperbolic layer:
//   out = exp_map_zero( log_map_zero(x) @ W^T + b )
// log_map_zero(x) = s(x_row) * x (per-row scalar) -> folded into epilogue.
// GEMM in bf16 WMMA (16x16x32), f32 accumulate. Whole W lives in LDS,
// staged once per block by the Tensor Data Mover with HW-inserted row
// padding (512B row + 16B pad = 528B stride -> bank-rotated ds_load_b128).
// ---------------------------------------------------------------------------
__global__ __launch_bounds__(256)
void hyperbolic_layer_kernel(const float* __restrict__ x,
                             const __bf16* __restrict__ Wb,
                             const float* __restrict__ b,
                             float* __restrict__ out)
{
    constexpr int D    = 256;              // d_in == d_out
    constexpr int KT   = 32;               // K tile (bf16 WMMA depth)
    constexpr int LDW  = 264;              // padded LDS row stride (bf16): 528B

    __shared__ __bf16 ldsW[256 * LDW];     // 132 KB: full W as bf16, padded

    const int tid  = threadIdx.x;
    const int wave = tid >> 5;
    const int lane = tid & 31;
    const int lr   = lane & 15;            // 0..15
    const int kh   = lane >> 4;            // 0/1 : lane half selects K block
    const int m0   = blockIdx.x * 128 + wave * 16;   // this wave's 16 rows
    const int row  = m0 + lr;

    // ---- one TDM DMA stages all of W (bf16) into LDS with row padding ----
    if (__builtin_amdgcn_readfirstlane(wave) == 0) {
        const unsigned long long ga = (unsigned long long)(uintptr_t)Wb;
        const unsigned int lds_base = (unsigned int)(uintptr_t)&ldsW[0];
        v4u g0;
        g0[0] = 1u;                                   // count=1 (valid D#)
        g0[1] = lds_base;                             // lds_addr (bytes)
        g0[2] = (unsigned int)ga;                     // global_addr[31:0]
        g0[3] = (unsigned int)((ga >> 32) & 0x1FFFFFFu) | (2u << 30); // type=2
        v8i g1;
        g1[0] = (int)((1u << 16)       // data_size = 2 bytes
                    | (1u << 20)       // pad_enable
                    | (6u << 22)       // pad_interval: 128 DWORDs (one row)
                    | (3u << 25));     // pad_amount: 4 DWORDs (16 B)
        g1[1] = (int)(256u << 16);     // tensor_dim0 = 256 (bits 48..79)
        g1[2] = (int)(256u << 16);     // tensor_dim1 = 256 (bits 80..111)
        g1[3] = (int)(256u << 16);     // tile_dim0   = 256 (bits 112..127)
        g1[4] = 256;                   // tile_dim1   = 256 (bits 128..143)
        g1[5] = 256;                   // tensor_dim0_stride = 256 (bits 160..)
        g1[6] = 0;
        g1[7] = 0;
        v4i z4 = {0, 0, 0, 0};         // groups 2/3: unused (2D tensor)
        v8i z8 = {0, 0, 0, 0, 0, 0, 0, 0};
        __builtin_amdgcn_tensor_load_to_lds(g0, g1, z4, z4, z8, 0);
        __builtin_amdgcn_s_wait_tensorcnt(0);
    }
    __syncthreads();

    v8f acc[16];
    #pragma unroll
    for (int f = 0; f < 16; ++f) { v8f z = {}; acc[f] = z; }

    float sumsq = 0.0f;                // partial sum of x^2 for row `row`

    for (int k0 = 0; k0 < D; k0 += KT) {
        // ---- A fragment (16x32 bf16 layout): lane half kh covers
        //      K = k0+kh*8+[0..7] (VGPR 0-3) and k0+16+kh*8+[0..7] (VGPR 4-7)
        const float* xp = x + row * D + k0 + kh * 8;
        v4f a0 = *(const v4f*)(xp);
        v4f a1 = *(const v4f*)(xp + 4);
        v4f a2 = *(const v4f*)(xp + 16);
        v4f a3 = *(const v4f*)(xp + 20);
        float av[16] = { a0.x, a0.y, a0.z, a0.w, a1.x, a1.y, a1.z, a1.w,
                         a2.x, a2.y, a2.z, a2.w, a3.x, a3.y, a3.z, a3.w };
        v16bf afrag;
        #pragma unroll
        for (int i = 0; i < 16; ++i) {
            afrag[i] = (__bf16)av[i];
            sumsq += av[i] * av[i];    // lanes L and L^16 jointly cover row
        }

        // ---- 16 column fragments: B(32x16) element e = W[n0+lr][k0+kh*8+e] (e<8), W[n0+lr][k0+16+kh*8+e-8] (e>=8)
        #pragma unroll
        for (int f = 0; f < 16; ++f) {
            const __bf16* bp = &ldsW[(f * 16 + lr) * LDW + k0 + kh * 8];
            v8bf blo = *(const v8bf*)bp;
            v8bf bhi = *(const v8bf*)(bp + 16);
            v16bf bfrag = __builtin_shufflevector(blo, bhi,
                0,1,2,3,4,5,6,7,8,9,10,11,12,13,14,15);
            acc[f] = __builtin_amdgcn_wmma_f32_16x16x32_bf16(
                false, afrag, false, bfrag, (short)0, acc[f], false, false);
        }
    }

    // ---- epilogue ----
    // complete ||x_row||: lanes L and L^16 hold the two disjoint K halves
    sumsq += __shfl_xor(sumsq, 16, 32);
    float xn = fmaxf(sqrtf(sumsq), EPSV);
    float u  = fminf(SQRT_C * xn, 1.0f - EPSV);
    float s  = (0.5f * logf((1.0f + u) / (1.0f - u))) / (SQRT_C * xn); // row lr

    float bv[16];
    #pragma unroll
    for (int f = 0; f < 16; ++f) bv[f] = b[f * 16 + lr];

    const float maxn = 1.0f / SQRT_C - 2.0f * EPSV;  // max_norm - EPS

    #pragma unroll
    for (int p = 0; p < 8; ++p) {
        const int r = p + 8 * kh;                // accum row within 16-strip
        float sp = __shfl(s, r, 32);             // row scale for this VGPR row
        float rq = 0.0f;
        #pragma unroll
        for (int f = 0; f < 16; ++f) {
            float t = sp * acc[f][p] + bv[f];    // tangent @ W^T + b
            acc[f][p] = t;
            rq += t * t;
        }
        // row-norm^2: sum across the 16 lanes of this half (cols)
        rq += __shfl_xor(rq, 1, 32);
        rq += __shfl_xor(rq, 2, 32);
        rq += __shfl_xor(rq, 4, 32);
        rq += __shfl_xor(rq, 8, 32);

        float vn = fmaxf(sqrtf(rq), EPSV);
        float th = tanhf(SQRT_C * vn);
        float g  = th / (SQRT_C * vn);           // exp_map scale
        float rn = fmaxf(th / SQRT_C, EPSV);     // ||res||
        float F  = g * fminf(rn, maxn) / rn;     // + project_to_ball scale

        float* op = out + (m0 + r) * D + lr;
        #pragma unroll
        for (int f = 0; f < 16; ++f) op[f * 16] = F * acc[f][p];
    }
}

extern "C" void kernel_launch(void* const* d_in, const int* in_sizes, int n_in,
                              void* d_out, int out_size, void* d_ws, size_t ws_size,
                              hipStream_t stream) {
    const float* x = (const float*)d_in[0];
    const float* W = (const float*)d_in[1];
    const float* b = (const float*)d_in[2];
    float* out = (float*)d_out;
    __bf16* Wb = (__bf16*)d_ws;        // 128 KB scratch for bf16 W

    // 1) convert W to bf16 once (65536 elems, 4 per thread)
    convert_w_kernel<<<64, 256, 0, stream>>>(W, Wb);

    // 2) fused GEMM + hyperbolic maps
    const int N = in_sizes[0] / 256;   // 262144 rows
    const int grid = N / 128;          // 128 rows per block
    hyperbolic_layer_kernel<<<grid, 256, 0, stream>>>(x, Wb, b, out);
}
